// SelfAttention_33621003994010
// MI455X (gfx1250) — matrix operand order repaired and checked by reading, not compiled, not run on hardware
//
#include <hip/hip_runtime.h>
#include <hip/hip_bf16.h>

typedef __attribute__((ext_vector_type(16))) __bf16 v16bf;
typedef __attribute__((ext_vector_type(8)))  float  v8f;
typedef __attribute__((ext_vector_type(4)))  int    v4i;

#define DIM 768
#define NH 12
#define HD 64
#define SEQ 2048
#define BATCH 2
#define TOKENS (BATCH*SEQ)     /* 4096 */
#define QKVDIM (3*DIM)         /* 2304 */

/* ---------- helpers ---------- */

static __device__ __forceinline__ unsigned short f2bf(float f) {
    union { float f; unsigned u; } a; a.f = f;
    unsigned u = a.u;
    if ((u & 0x7fffffffu) > 0x7f800000u) return (unsigned short)((u >> 16) | 0x0040u); // NaN
    unsigned r = u + 0x7fffu + ((u >> 16) & 1u);  // round-to-nearest-even
    return (unsigned short)(r >> 16);
}

union AF      { v16bf v; unsigned u[8]; };   // A fragment (16 consecutive K per lane half)
union BF16x16 { v16bf v; uint4 q[2];   };    // B fragment: 16 consecutive K per lane

static __device__ __forceinline__ v8f vzero8() {
    v8f z = {0.f,0.f,0.f,0.f,0.f,0.f,0.f,0.f};
    return z;
}

// A-fragment u32 index for dword i, lane-half kh (16 consecutive K per lane half)
static __device__ __forceinline__ int a_u32_idx(int i, int kh) {
    return i + 8*kh;
}

/* ---------- CDNA5 async global->LDS copy (ASYNCcnt-tracked), with fallback ---------- */

#if __has_builtin(__builtin_amdgcn_global_load_async_to_lds_b128)
#define HAVE_ASYNC_LDS 1
typedef __attribute__((address_space(1))) v4i gv4i;   // global (AS1) int4
typedef __attribute__((address_space(3))) v4i lv4i;   // LDS (AS3) int4
static __device__ __forceinline__ void async_cp16(void* lds, const void* g) {
    // flat shared address: low 32 bits == LDS byte offset (ISA 10.2 aperture rules)
    __builtin_amdgcn_global_load_async_to_lds_b128(
        (gv4i*)(unsigned long long)g,
        (lv4i*)(unsigned)(unsigned long long)lds, 0, 0);
}
static __device__ __forceinline__ void wait_async() {
#if __has_builtin(__builtin_amdgcn_s_wait_asynccnt)
    __builtin_amdgcn_s_wait_asynccnt(0);
#else
    asm volatile("s_wait_asynccnt 0x0" ::: "memory");
#endif
}
#else
#define HAVE_ASYNC_LDS 0
static __device__ __forceinline__ void async_cp16(void* lds, const void* g) {
    *(uint4*)lds = *(const uint4*)g;
}
static __device__ __forceinline__ void wait_async() {}
#endif

/* ---------- fp32 -> bf16 convert ---------- */

__global__ __launch_bounds__(256) void cvt_f32_bf16(const float* __restrict__ in,
                                                    unsigned short* __restrict__ out, int n) {
    int i = blockIdx.x * blockDim.x + threadIdx.x;
    if (i < n) out[i] = f2bf(in[i]);
}

/* ---------- tiled WMMA GEMM: C[M,N] = A[M,K](bf16) * B[K,N](bf16) + bias ----------
   block tile 128x128, 8 waves (4M x 2N), each wave 32x64 = 2x4 WMMA tiles.
   Double-buffered LDS; A tile staged via async-to-LDS, B tile transposed manually
   (global loads issued before compute, ds_stores after compute). */

template <bool F32OUT>
__global__ __launch_bounds__(256)
void gemm_bf16_wmma(const unsigned short* __restrict__ A,   // [M][K] bf16 bits
                    const unsigned short* __restrict__ B,   // [K][N] bf16 bits
                    const float* __restrict__ bias,         // [N]
                    void* __restrict__ Cout,                // bf16[M][N] or f32[M][N]
                    int M, int N, int K)
{
    __shared__ __attribute__((aligned(32))) unsigned short lA[2][128 * 32]; // [m][k]
    __shared__ __attribute__((aligned(32))) unsigned short lB[2][128 * 32]; // [n][k]

    const int tid  = threadIdx.x;
    const int lane = tid & 31;
    const int w    = tid >> 5;
    const int wm   = w >> 1;        // 0..3  (M offset 32*wm)
    const int wn   = w & 1;         // 0..1  (N offset 64*wn)
    const int kh   = lane >> 4;
    const int l16  = lane & 15;

    const int bm = blockIdx.y * 128;
    const int bn = blockIdx.x * 128;

    // this thread's two staging chunks (16B each)
    const int ar0 = tid >> 2,         aj0 = (tid & 3) * 8;         // A chunk 0
    const int ar1 = (tid + 256) >> 2, aj1 = aj0;                   // A chunk 1
    const int bk0 = tid >> 4,         bj0 = (tid & 15) * 8;        // B chunk 0
    const int bk1 = (tid + 256) >> 4, bj1 = bj0;                   // B chunk 1

    v8f acc[2][4];
#pragma unroll
    for (int ms = 0; ms < 2; ++ms)
#pragma unroll
        for (int ns = 0; ns < 4; ++ns) acc[ms][ns] = vzero8();

    // ---- prologue: stage k-step 0 into buffer 0 ----
    async_cp16(&lA[0][ar0 * 32 + aj0], A + (size_t)(bm + ar0) * K + aj0);
    async_cp16(&lA[0][ar1 * 32 + aj1], A + (size_t)(bm + ar1) * K + aj1);
    {
        uint4 s0 = *(const uint4*)(B + (size_t)bk0 * N + bn + bj0);
        uint4 s1 = *(const uint4*)(B + (size_t)bk1 * N + bn + bj1);
        const unsigned short* p0 = (const unsigned short*)&s0;
        const unsigned short* p1 = (const unsigned short*)&s1;
#pragma unroll
        for (int i = 0; i < 8; ++i) {
            lB[0][(bj0 + i) * 32 + bk0] = p0[i];
            lB[0][(bj1 + i) * 32 + bk1] = p1[i];
        }
    }
    wait_async();
    __syncthreads();

    const int T = K >> 5;
    for (int t = 0; t < T; ++t) {
        const int buf = t & 1;
        const bool pre = (t + 1 < T);
        uint4 s0, s1;
        if (pre) {
            const int k1 = (t + 1) << 5;
            // async A stage for t+1 (no VGPR data, overlaps compute)
            async_cp16(&lA[buf ^ 1][ar0 * 32 + aj0], A + (size_t)(bm + ar0) * K + k1 + aj0);
            async_cp16(&lA[buf ^ 1][ar1 * 32 + aj1], A + (size_t)(bm + ar1) * K + k1 + aj1);
            // B global loads issued now, stored to LDS after compute
            s0 = *(const uint4*)(B + (size_t)(k1 + bk0) * N + bn + bj0);
            s1 = *(const uint4*)(B + (size_t)(k1 + bk1) * N + bn + bj1);
        }

        // ---- compute on buffer buf ----
        AF a[2];
#pragma unroll
        for (int ms = 0; ms < 2; ++ms) {
            const unsigned* base = (const unsigned*)lA[buf] + (wm * 32 + ms * 16 + l16) * 16;
#pragma unroll
            for (int i = 0; i < 8; ++i) a[ms].u[i] = base[a_u32_idx(i, kh)];
        }
#pragma unroll
        for (int ns = 0; ns < 4; ++ns) {
            BF16x16 b;
            const uint4* bp = (const uint4*)(lB[buf] + (wn * 64 + ns * 16 + l16) * 32 + 16 * kh);
            b.q[0] = bp[0]; b.q[1] = bp[1];
#pragma unroll
            for (int ms = 0; ms < 2; ++ms)
                acc[ms][ns] = __builtin_amdgcn_wmma_f32_16x16x32_bf16(
                    false, a[ms].v, false, b.v, (short)0, acc[ms][ns], false, false);
        }

        if (pre) {
            const unsigned short* p0 = (const unsigned short*)&s0;
            const unsigned short* p1 = (const unsigned short*)&s1;
#pragma unroll
            for (int i = 0; i < 8; ++i) {
                lB[buf ^ 1][(bj0 + i) * 32 + bk0] = p0[i];
                lB[buf ^ 1][(bj1 + i) * 32 + bk1] = p1[i];
            }
        }
        wait_async();
        __syncthreads();
    }

    // epilogue: bias + store (C layout: dword g -> row g + 8*kh, col = lane&15)
#pragma unroll
    for (int ms = 0; ms < 2; ++ms)
#pragma unroll
        for (int ns = 0; ns < 4; ++ns) {
            int col = bn + wn * 64 + ns * 16 + l16;
            float bv = bias[col];
#pragma unroll
            for (int g = 0; g < 8; ++g) {
                int row = bm + wm * 32 + ms * 16 + g + 8 * kh;
                float v = acc[ms][ns][g] + bv;
                if (F32OUT) ((float*)Cout)[(size_t)row * N + col] = v;
                else ((unsigned short*)Cout)[(size_t)row * N + col] = f2bf(v);
            }
        }
}

/* ---------- flash attention forward (bf16 in/out, fp32 softmax) ----------
   grid (SEQ/128, BATCH*NH), block 256, each wave owns 16 queries.
   Double-buffered 32-key K/V tiles: K staged async-to-LDS, V transposed manually
   with loads before / stores after the WMMA block. */

__global__ __launch_bounds__(256)
void attn_fwd(const unsigned short* __restrict__ qkv,   // [TOKENS][2304] bf16
              unsigned short* __restrict__ y)           // [TOKENS][768]  bf16
{
    __shared__ __attribute__((aligned(32))) unsigned short lK[2][32 * 64];     // [key][hd]
    __shared__ __attribute__((aligned(32))) unsigned short lV[2][64 * 32];     // [hd][key]
    __shared__ __attribute__((aligned(32))) unsigned short lP[8 * 16 * 32];    // per-wave P

    const int tid  = threadIdx.x;
    const int lane = tid & 31;
    const int w    = tid >> 5;
    const int kh   = lane >> 4;
    const int l16  = lane & 15;

    const int bb = blockIdx.y / NH;
    const int hh = blockIdx.y % NH;
    const int q0 = blockIdx.x * 128 + w * 16;
    const size_t tokbase = (size_t)bb * SEQ;
    const float scale = 0.125f;   // 1/sqrt(64)

    const int stok = tid >> 3;          // staging: token-in-block
    const int sj   = (tid & 7) * 8;     // staging: hd chunk

    // preload Q A-fragments for hd chunks [0,32) and [32,64)
    AF qa[2];
    {
        const unsigned* qp = (const unsigned*)(qkv + (tokbase + q0 + l16) * QKVDIM + hh * HD);
#pragma unroll
        for (int c = 0; c < 2; ++c)
#pragma unroll
            for (int i = 0; i < 8; ++i)
                qa[c].u[i] = qp[16 * c + a_u32_idx(i, kh)];
    }

    v8f o[4];
#pragma unroll
    for (int nt = 0; nt < 4; ++nt) o[nt] = vzero8();
    float m[8], l[8];
#pragma unroll
    for (int g = 0; g < 8; ++g) { m[g] = -3.0e38f; l[g] = 0.f; }

    // ---- prologue: stage key-block 0 into buffer 0 ----
    {
        const unsigned short* row = qkv + (tokbase + stok) * QKVDIM + hh * HD;
        async_cp16(&lK[0][stok * 64 + sj], row + DIM + sj);
        uint4 vv = *(const uint4*)(row + 2 * DIM + sj);
        const unsigned short* s = (const unsigned short*)&vv;
#pragma unroll
        for (int i = 0; i < 8; ++i) lV[0][(sj + i) * 32 + stok] = s[i];
    }
    wait_async();
    __syncthreads();

    const int NT = SEQ / 32;   // 64
    for (int t = 0; t < NT; ++t) {
        const int buf = t & 1;
        const bool pre = (t + 1 < NT);
        uint4 vv;
        if (pre) {
            const unsigned short* row = qkv + (tokbase + (t + 1) * 32 + stok) * QKVDIM + hh * HD;
            async_cp16(&lK[buf ^ 1][stok * 64 + sj], row + DIM + sj);
            vv = *(const uint4*)(row + 2 * DIM + sj);
        }

        // S = Q * K^T for two 16-key tiles
        v8f s0 = vzero8(), s1 = vzero8();
#pragma unroll
        for (int c = 0; c < 2; ++c) {
            BF16x16 bk;
            const uint4* p0 = (const uint4*)(lK[buf] + l16 * 64 + c * 32 + 16 * kh);
            bk.q[0] = p0[0]; bk.q[1] = p0[1];
            s0 = __builtin_amdgcn_wmma_f32_16x16x32_bf16(false, qa[c].v, false, bk.v,
                                                         (short)0, s0, false, false);
            const uint4* p1 = (const uint4*)(lK[buf] + (16 + l16) * 64 + c * 32 + 16 * kh);
            bk.q[0] = p1[0]; bk.q[1] = p1[1];
            s1 = __builtin_amdgcn_wmma_f32_16x16x32_bf16(false, qa[c].v, false, bk.v,
                                                         (short)0, s1, false, false);
        }

        // online softmax (C-layout row lives in one VGPR across a 16-lane half)
        float mb[8], p0a[8], p1a[8], rs[8], al[8];
#pragma unroll
        for (int g = 0; g < 8; ++g) {
            float v0 = s0[g] * scale, v1 = s1[g] * scale;
            s0[g] = v0; s1[g] = v1;
            mb[g] = fmaxf(v0, v1);
        }
#pragma unroll
        for (int d = 1; d < 16; d <<= 1)
#pragma unroll
            for (int g = 0; g < 8; ++g)
                mb[g] = fmaxf(mb[g], __shfl_xor(mb[g], d, 32));
#pragma unroll
        for (int g = 0; g < 8; ++g) {
            float mn = fmaxf(m[g], mb[g]);
            al[g] = __expf(m[g] - mn);
            m[g]  = mn;
            p0a[g] = __expf(s0[g] - mn);
            p1a[g] = __expf(s1[g] - mn);
            rs[g] = p0a[g] + p1a[g];
        }
#pragma unroll
        for (int d = 1; d < 16; d <<= 1)
#pragma unroll
            for (int g = 0; g < 8; ++g)
                rs[g] += __shfl_xor(rs[g], d, 32);
#pragma unroll
        for (int g = 0; g < 8; ++g) l[g] = l[g] * al[g] + rs[g];

        // rescale O by alpha (running sum rides the WMMA C operand)
#pragma unroll
        for (int nt = 0; nt < 4; ++nt)
#pragma unroll
            for (int g = 0; g < 8; ++g) o[nt][g] *= al[g];

        // restage P (C layout) -> LDS -> A-fragment layout (intra-wave, DS in-order)
        unsigned short* pw = lP + w * 512;
#pragma unroll
        for (int g = 0; g < 8; ++g) {
            int row = g + 8 * kh;
            pw[row * 32 + l16]      = f2bf(p0a[g]);
            pw[row * 32 + 16 + l16] = f2bf(p1a[g]);
        }
        AF pa;
        const unsigned* pb = (const unsigned*)pw + l16 * 16;
#pragma unroll
        for (int i = 0; i < 8; ++i) pa.u[i] = pb[a_u32_idx(i, kh)];

        // O += P * V
#pragma unroll
        for (int nt = 0; nt < 4; ++nt) {
            BF16x16 bv;
            const uint4* vp = (const uint4*)(lV[buf] + (nt * 16 + l16) * 32 + 16 * kh);
            bv.q[0] = vp[0]; bv.q[1] = vp[1];
            o[nt] = __builtin_amdgcn_wmma_f32_16x16x32_bf16(false, pa.v, false, bv.v,
                                                            (short)0, o[nt], false, false);
        }

        if (pre) {
            const unsigned short* s = (const unsigned short*)&vv;
#pragma unroll
            for (int i = 0; i < 8; ++i) lV[buf ^ 1][(sj + i) * 32 + stok] = s[i];
        }
        wait_async();
        __syncthreads();
    }

    // normalize and store y (bf16)
#pragma unroll
    for (int nt = 0; nt < 4; ++nt)
#pragma unroll
        for (int g = 0; g < 8; ++g) {
            int row = q0 + g + 8 * kh;
            int col = hh * HD + nt * 16 + l16;
            y[(tokbase + row) * DIM + col] = f2bf(o[nt][g] / l[g]);
        }
}

/* ---------- host launcher ---------- */

extern "C" void kernel_launch(void* const* d_in, const int* in_sizes, int n_in,
                              void* d_out, int out_size, void* d_ws, size_t ws_size,
                              hipStream_t stream) {
    const float* x     = (const float*)d_in[0];   // [2,2048,768]
    const float* Wqkv  = (const float*)d_in[1];   // [768,2304]
    const float* bqkv  = (const float*)d_in[2];   // [2304]
    const float* Wproj = (const float*)d_in[3];   // [768,768]
    const float* bproj = (const float*)d_in[4];   // [768]
    float* out = (float*)d_out;                   // [2,2048,768]

    unsigned short* xb   = (unsigned short*)d_ws;                 // 4096*768
    unsigned short* wqb  = xb  + (size_t)TOKENS * DIM;            // 768*2304
    unsigned short* wpb  = wqb + (size_t)DIM * QKVDIM;            // 768*768
    unsigned short* qkvb = wpb + (size_t)DIM * DIM;               // 4096*2304
    unsigned short* yb   = qkvb + (size_t)TOKENS * QKVDIM;        // 4096*768

    const int n_x  = TOKENS * DIM;
    const int n_wq = DIM * QKVDIM;
    const int n_wp = DIM * DIM;
    cvt_f32_bf16<<<(n_x  + 255) / 256, 256, 0, stream>>>(x,     xb,  n_x);
    cvt_f32_bf16<<<(n_wq + 255) / 256, 256, 0, stream>>>(Wqkv,  wqb, n_wq);
    cvt_f32_bf16<<<(n_wp + 255) / 256, 256, 0, stream>>>(Wproj, wpb, n_wp);

    // qkv = x @ W_qkv + b_qkv   (bf16 out)
    gemm_bf16_wmma<false><<<dim3(QKVDIM / 128, TOKENS / 128), 256, 0, stream>>>(
        xb, wqb, bqkv, qkvb, TOKENS, QKVDIM, DIM);

    // flash attention per (b,h)
    attn_fwd<<<dim3(SEQ / 128, BATCH * NH), 256, 0, stream>>>(qkvb, yb);

    // out = y @ W_proj + b_proj   (fp32 out)
    gemm_bf16_wmma<true><<<dim3(DIM / 128, TOKENS / 128), 256, 0, stream>>>(
        yb, wpb, bproj, out, TOKENS, DIM, DIM);
}
